// GraphAttentionLayer_7713761263781
// MI455X (gfx1250) — matrix operand order repaired and checked by reading, not compiled, not run on hardware
//
#include <hip/hip_runtime.h>
#include <hip/hip_bf16.h>

// ---------------------------------------------------------------------------
// GraphAttentionLayer forward for MI455X (gfx1250), wave32 + WMMA bf16 + TDM.
//
//   hp = h @ W                      (bf16 WMMA, f32 acc)
//   e  = hp @ hp^T  (never stored)  (bf16 WMMA, streamed twice, flash-style)
//   softmax over axis=1 -> column stats M[m], S[m]
//   row_mass[n] = sum_m exp(lrelu(e[n,m])-M[m])/S[m];  out = hp * row_mass
//
// Streaming passes stage 16KB chunks of hp_bf16 into LDS with the Tensor
// Data Mover (tensor_load_to_lds, TENSORcnt), double-buffered, shared by the
// 8 waves of each block.
// ---------------------------------------------------------------------------

typedef __attribute__((ext_vector_type(16))) __bf16 v16bf;
typedef __attribute__((ext_vector_type(8)))  __bf16 v8bf;
typedef __attribute__((ext_vector_type(8)))  float  v8f;
typedef __attribute__((ext_vector_type(4)))  unsigned int u32x4;
typedef __attribute__((ext_vector_type(8)))  int i32x8;
typedef __attribute__((ext_vector_type(4)))  int i32x4;

#define ALPHA  0.2f
#define LOG2E  1.4426950408889634f
#define BATCH  4
#define NN     4096
#define FIN    256
#define FOUT   128
#define ROWS   (BATCH * NN)            // 16384
#define GROUP  4                       // 16-row tiles per staged chunk
#define CHUNK_BYTES (GROUP * 16 * FOUT * 2)   // 16 KB
#define NCHUNK (NN / (16 * GROUP))     // 64 chunks per batch

static __device__ __forceinline__ v8f wmma_bf16(v16bf a, v16bf b, v8f c) {
  return __builtin_amdgcn_wmma_f32_16x16x32_bf16(
      false, a, false, b, (short)0, c, false, false);
}

static __device__ __forceinline__ float half16_sum(float x) {
  x += __shfl_xor(x, 1, 16);
  x += __shfl_xor(x, 2, 16);
  x += __shfl_xor(x, 4, 16);
  x += __shfl_xor(x, 8, 16);
  return x;
}

// 1-D contiguous TDM copy global -> LDS, 8-byte elements (nbytes % 8 == 0).
// D# built per CDNA5 ISA 8.3-8.6; remaining groups zero (<=2D tensor).
// This toolchain exposes the 6-arg builtin form.
static __device__ __forceinline__ void tdm_load_1d(unsigned int lds_off,
                                                   const void* gptr,
                                                   unsigned int nbytes) {
  const unsigned long long ga = (unsigned long long)(uintptr_t)gptr;
  const unsigned int ne = nbytes >> 3;             // 8B elements
  u32x4 g0;
  g0[0] = 1u;                                      // count=1 (valid user D#)
  g0[1] = lds_off;                                 // lds_addr
  g0[2] = (unsigned int)ga;                        // global_addr[31:0]
  g0[3] = (unsigned int)((ga >> 32) & 0x01FFFFFFu) // global_addr[56:32]
          | (2u << 30);                            // type=2 ("image")
  i32x8 g1;
  g1[0] = (int)(3u << 16);                         // data_size=3 (8B), mask=0
  g1[1] = (int)((ne & 0xFFFFu) << 16);             // tensor_dim0[15:0]
  g1[2] = (int)(((ne >> 16) & 0xFFFFu)             // tensor_dim0[31:16]
          | (1u << 16));                           // tensor_dim1 = 1
  g1[3] = (int)((ne & 0xFFFFu) << 16);             // tile_dim0 = ne
  g1[4] = 1;                                       // tile_dim1=1, tile_dim2=0
  g1[5] = (int)ne;                                 // tensor_dim0_stride[31:0]
  g1[6] = 0;                                       // stride hi / dim1_stride
  g1[7] = 0;
  i32x4 g2 = {0, 0, 0, 0};
  i32x4 g3 = {0, 0, 0, 0};
  i32x8 g4 = {0, 0, 0, 0, 0, 0, 0, 0};
  __builtin_amdgcn_tensor_load_to_lds(g0, g1, g2, g3, g4, 0);
}

// ---------------------------------------------------------------------------
// Kernel 1: hp = h @ W.  One wave per 16x16 output tile, K=256 in 8 WMMA ops.
// ---------------------------------------------------------------------------
__global__ __launch_bounds__(256) void k_gemm_hp(
    const float* __restrict__ h, const float* __restrict__ W,
    float* __restrict__ hpf, unsigned short* __restrict__ hpb_u) {
  __bf16* hpb = reinterpret_cast<__bf16*>(hpb_u);
  const int wave = (blockIdx.x * blockDim.x + threadIdx.x) >> 5;
  const int lane = threadIdx.x & 31;
  const int l15  = lane & 15;
  const int hi   = lane >> 4;
  const int rt = wave >> 3;            // 1024 row tiles
  const int ct = wave & 7;             // 8 col tiles
  const int r0 = rt * 16, c0 = ct * 16;

  v8f acc = {};
#pragma unroll
  for (int kc = 0; kc < 8; ++kc) {
    const int kb = kc * 32;
    // A (16x32 bf16): lane = row l15, K = hi*8+{0..7} and hi*8+{16..23}
    const float* pa = h + (size_t)(r0 + l15) * FIN + kb + hi * 8;
    v16bf a;
#pragma unroll
    for (int t = 0; t < 8; ++t) {
      a[t]     = (__bf16)pa[t];
      a[t + 8] = (__bf16)pa[t + 16];
    }
    // B (32x16 bf16): lane = col l15, K = hi*8+{0..7} and hi*8+{16..23}
    const float* pb = W + (size_t)(kb + hi * 8) * FOUT + c0 + l15;
    v16bf b;
#pragma unroll
    for (int t = 0; t < 16; ++t) b[t] = (__bf16)pb[(size_t)(t + (t & 8)) * FOUT];
    acc = wmma_bf16(a, b, acc);
  }
  const int col = c0 + l15;
#pragma unroll
  for (int r = 0; r < 8; ++r) {
    const int row = r0 + r + hi * 8;
    const float v = acc[r];
    hpf[(size_t)row * FOUT + col] = v;
    hpb[(size_t)row * FOUT + col] = (__bf16)v;
  }
}

// ---------------------------------------------------------------------------
// Kernel 2: column stats of f = leakyrelu(hp hp^T), online softmax.
// Wave owns a 16-column tile (B fixed in regs); block of 8 waves shares the
// streamed A rows via TDM-staged LDS chunks (double buffered).
// ---------------------------------------------------------------------------
__global__ __launch_bounds__(256) void k_col_stats(
    const unsigned short* __restrict__ hpb_u,
    float* __restrict__ Mcol, float* __restrict__ Sinv) {
  __shared__ __attribute__((aligned(32))) unsigned char smem[2 * CHUNK_BYTES];
  const __bf16* hpb = reinterpret_cast<const __bf16*>(hpb_u);
  const int tid  = threadIdx.x;
  const int wid  = tid >> 5;
  const int wave = blockIdx.x * 8 + wid;           // 0..1023
  const int lane = tid & 31, l15 = lane & 15, hi = lane >> 4;
  const int b    = wave >> 8;                      // 256 col tiles / batch
  const int gc0  = wave * 16;                      // global column base

  // B fixed: column j = l15 is hp row gc0+l15 (contiguous 32B chunks).
  const __bf16* pb = hpb + (size_t)(gc0 + l15) * FOUT;
  v16bf bm[4];
#pragma unroll
  for (int kc = 0; kc < 4; ++kc)
    bm[kc] = *reinterpret_cast<const v16bf*>(pb + kc * 32 + hi * 16);

  const unsigned char* gsrc =
      (const unsigned char*)(hpb + (size_t)b * NN * FOUT);  // batch rows
  const unsigned int lds0 = (unsigned int)(uintptr_t)&smem[0];

  if (wid == 0) tdm_load_1d(lds0, gsrc, CHUNK_BYTES);       // prologue

  float Mc = -3.0e38f, Sc = 0.0f;
  for (int c = 0; c < NCHUNK; ++c) {
    const int cb = c & 1;
    if (wid == 0) __builtin_amdgcn_s_wait_tensorcnt(0);
    __syncthreads();                                // chunk c in LDS for all
    if (wid == 0 && (c + 1) < NCHUNK)
      tdm_load_1d(lds0 + (1 - cb) * CHUNK_BYTES,
                  gsrc + (size_t)(c + 1) * CHUNK_BYTES, CHUNK_BYTES);

    const __bf16* abase = (const __bf16*)(smem + cb * CHUNK_BYTES);
#pragma unroll
    for (int t = 0; t < GROUP; ++t) {
      const __bf16* pa = abase + (size_t)(t * 16 + l15) * FOUT + hi * 16;
      v8f cc = {};
#pragma unroll
      for (int kc = 0; kc < 4; ++kc) {
        union { v16bf v; v8bf h2[2]; } av;
        av.h2[0] = *reinterpret_cast<const v8bf*>(pa + kc * 32);
        av.h2[1] = *reinterpret_cast<const v8bf*>(pa + kc * 32 + 8);
        cc = wmma_bf16(av.v, bm[kc], cc);
      }
      float f[8];
      float tmax = -3.0e38f;
#pragma unroll
      for (int r = 0; r < 8; ++r) {
        const float x = cc[r];
        f[r] = (x >= 0.0f) ? x : ALPHA * x;
        tmax = fmaxf(tmax, f[r]);
      }
      tmax = fmaxf(tmax, __shfl_xor(tmax, 16, 32));  // combine row halves
      const float newM  = fmaxf(Mc, tmax);
      const float scale = __builtin_amdgcn_exp2f((Mc - newM) * LOG2E);
      float ssum = 0.0f;
#pragma unroll
      for (int r = 0; r < 8; ++r)
        ssum += __builtin_amdgcn_exp2f((f[r] - newM) * LOG2E);
      ssum += __shfl_xor(ssum, 16, 32);
      Sc = Sc * scale + ssum;
      Mc = newM;
    }
  }
  if (hi == 0) {                        // halves hold identical column stats
    Mcol[gc0 + l15] = Mc;
    Sinv[gc0 + l15] = __builtin_amdgcn_rcpf(Sc);
  }
}

// ---------------------------------------------------------------------------
// Kernel 3: row_mass[n] = sum_m exp(f[n,m]-M[m]) * Sinv[m].
// Wave owns a 16-row tile (A fixed in regs); streamed B columns staged via
// TDM-shared LDS chunks as in kernel 2.
// ---------------------------------------------------------------------------
__global__ __launch_bounds__(256) void k_row_mass(
    const unsigned short* __restrict__ hpb_u,
    const float* __restrict__ Mcol, const float* __restrict__ Sinv,
    float* __restrict__ rowm) {
  __shared__ __attribute__((aligned(32))) unsigned char smem[2 * CHUNK_BYTES];
  const __bf16* hpb = reinterpret_cast<const __bf16*>(hpb_u);
  const int tid  = threadIdx.x;
  const int wid  = tid >> 5;
  const int wave = blockIdx.x * 8 + wid;           // 0..1023
  const int lane = tid & 31, l15 = lane & 15, hi = lane >> 4;
  const int b    = wave >> 8;
  const int gr0  = wave * 16;                      // global row base

  // A fixed: rows gr0..gr0+15
  const __bf16* pa0 = hpb + (size_t)(gr0 + l15) * FOUT + hi * 16;
  v16bf am[4];
#pragma unroll
  for (int kc = 0; kc < 4; ++kc) {
    union { v16bf v; v8bf h2[2]; } av;
    av.h2[0] = *reinterpret_cast<const v8bf*>(pa0 + kc * 32);
    av.h2[1] = *reinterpret_cast<const v8bf*>(pa0 + kc * 32 + 8);
    am[kc] = av.v;
  }

  float rm[8];
#pragma unroll
  for (int r = 0; r < 8; ++r) rm[r] = 0.0f;

  const int colbase = b * NN;
  const unsigned char* gsrc =
      (const unsigned char*)(hpb + (size_t)colbase * FOUT);
  const unsigned int lds0 = (unsigned int)(uintptr_t)&smem[0];

  if (wid == 0) tdm_load_1d(lds0, gsrc, CHUNK_BYTES);       // prologue

  for (int c = 0; c < NCHUNK; ++c) {
    const int cb = c & 1;
    if (wid == 0) __builtin_amdgcn_s_wait_tensorcnt(0);
    __syncthreads();
    if (wid == 0 && (c + 1) < NCHUNK)
      tdm_load_1d(lds0 + (1 - cb) * CHUNK_BYTES,
                  gsrc + (size_t)(c + 1) * CHUNK_BYTES, CHUNK_BYTES);

    const __bf16* bbase = (const __bf16*)(smem + cb * CHUNK_BYTES);
#pragma unroll
    for (int t = 0; t < GROUP; ++t) {
      const int gm = colbase + (c * GROUP + t) * 16 + l15;  // lane's column
      const __bf16* pbp = bbase + (size_t)(t * 16 + l15) * FOUT;
      v8f cc = {};
#pragma unroll
      for (int kc = 0; kc < 4; ++kc) {
        const v16bf bv =
            *reinterpret_cast<const v16bf*>(pbp + kc * 32 + hi * 16);
        cc = wmma_bf16(am[kc], bv, cc);
      }
      const float Mc = Mcol[gm];
      const float Iv = Sinv[gm];
#pragma unroll
      for (int r = 0; r < 8; ++r) {
        const float x = cc[r];
        const float f = (x >= 0.0f) ? x : ALPHA * x;
        rm[r] += __builtin_amdgcn_exp2f((f - Mc) * LOG2E) * Iv;
      }
    }
  }
#pragma unroll
  for (int r = 0; r < 8; ++r) rm[r] = half16_sum(rm[r]);
  if (l15 == 0) {                      // lane 0 -> rows 0..7, lane 16 -> 8..15
#pragma unroll
    for (int r = 0; r < 8; ++r) rowm[gr0 + r + hi * 8] = rm[r];
  }
}

// ---------------------------------------------------------------------------
// Kernel 4: out = hp_f32 * row_mass (broadcast over FOUT=128 columns).
// ---------------------------------------------------------------------------
__global__ __launch_bounds__(256) void k_scale(
    const float* __restrict__ hpf, const float* __restrict__ rowm,
    float* __restrict__ out) {
  const int i = blockIdx.x * blockDim.x + threadIdx.x;   // exact grid
  out[i] = hpf[i] * rowm[i >> 7];                        // FOUT == 128
}

// ---------------------------------------------------------------------------
extern "C" void kernel_launch(void* const* d_in, const int* in_sizes, int n_in,
                              void* d_out, int out_size, void* d_ws, size_t ws_size,
                              hipStream_t stream) {
  (void)in_sizes; (void)n_in; (void)out_size; (void)ws_size;
  const float* h = (const float*)d_in[0];   // [B,N,Fin] f32
  // d_in[1] = adj (unused by forward)
  const float* W = (const float*)d_in[2];   // [Fin,Fout] f32
  float* out = (float*)d_out;               // [B,N,Fout] f32

  char* ws = (char*)d_ws;                   // ~12.2 MB total
  float*          hpf  = (float*)ws;                                   // 8 MB
  unsigned short* hpb  = (unsigned short*)(ws + (size_t)ROWS*FOUT*4);  // 4 MB
  float*          Mcol = (float*)(ws + (size_t)ROWS*FOUT*6);           // 64 KB
  float*          Sinv = Mcol + ROWS;                                  // 64 KB
  float*          rowm = Sinv + ROWS;                                  // 64 KB

  k_gemm_hp  <<<1024, 256, 0, stream>>>(h, W, hpf, hpb);
  k_col_stats<<< 128, 256, 0, stream>>>(hpb, Mcol, Sinv);
  k_row_mass <<< 128, 256, 0, stream>>>(hpb, Mcol, Sinv, rowm);
  k_scale    <<<(ROWS * FOUT) / 256, 256, 0, stream>>>(hpf, rowm, out);
}
